// GAT_69578470195575
// MI455X (gfx1250) — matrix operand order repaired and checked by reading, not compiled, not run on hardware
//
#include <hip/hip_runtime.h>

// ---------------- problem constants (match reference) ----------------
static constexpr int NN   = 50000;   // nodes
static constexpr int EE   = 800000;  // edges (before self loops)
static constexpr int ETOT = EE + NN; // with self loops
static constexpr int HH   = 4;       // heads
static constexpr int CC   = 64;      // channels per head
static constexpr int HC   = HH * CC; // 256
static constexpr int FIN  = 128;     // input features

typedef __attribute__((ext_vector_type(16))) __bf16 v16bf;
typedef __attribute__((ext_vector_type(8)))  float  v8f;

union FragBF { unsigned int u[8]; uint4 q[2]; v16bf v; };

// ---------------- helpers ----------------
__device__ inline unsigned short f2bf(float f) {
    unsigned u = __float_as_uint(f);
    unsigned r = u + 0x7FFFu + ((u >> 16) & 1u);   // round-to-nearest-even
    return (unsigned short)(r >> 16);
}

__device__ inline void atomicMaxF(float* addr, float val) {
    // sign-punned float max: correct for all non-NaN floats
    if (val >= 0.f) atomicMax((int*)addr, __float_as_int(val));
    else            atomicMin((unsigned int*)addr, __float_as_uint(val));
}

// ---------------- elementwise utility kernels ----------------
__global__ void gat_f32_to_bf16(const float* __restrict__ src,
                                unsigned short* __restrict__ dst, int n) {
    int gid = blockIdx.x * blockDim.x + threadIdx.x;
    if (gid < n) dst[gid] = f2bf(src[gid]);
}

__global__ void gat_fill(float* __restrict__ p, float v, int n) {
    int gid = blockIdx.x * blockDim.x + threadIdx.x;
    if (gid < n) p[gid] = v;
}

// ---------------- pack weights into WMMA B-fragment layout ----------------
// B fragment (16-bit, 32x16): lane holds col = lane&15, K = {kb2..kb2+15}, kb2=(lane>>4)*16.
// Packed: P[((nt*ksteps + ks)*32 + lane)*16 + i] = W[ks*32 + kb2 + i][nt*16 + col] (bf16)
// -> GEMM reads each fragment as two contiguous uint4 loads.
__global__ void gat_pack_B(const float* __restrict__ W,
                           unsigned short* __restrict__ P, int K, int Nc) {
    int tid = blockIdx.x * blockDim.x + threadIdx.x;
    const int ksteps = K >> 5;
    const int total = (Nc >> 4) * ksteps * 32;
    if (tid >= total) return;
    int lane = tid & 31;
    int rem  = tid >> 5;
    int ks   = rem % ksteps;
    int nt   = rem / ksteps;
    int col  = nt * 16 + (lane & 15);
    int kb   = ks * 32 + (lane >> 4) * 16;
    unsigned short* dst = P + (size_t)tid * 16;
#pragma unroll
    for (int i = 0; i < 16; ++i)
        dst[i] = f2bf(W[(size_t)(kb + i) * Nc + col]);
}

// ---------------- WMMA bf16 GEMM: C[M,Nc] = A[M,K] * B[K,Nc] ----------------
// A row-major bf16; B pre-packed fragments; C f32.
// One wave -> 16x64 output strip (4 accumulators), reusing one A fragment per k-step.
__global__ __launch_bounds__(256)
void gat_gemm_bf16_wmma(const unsigned short* __restrict__ A,
                        const unsigned short* __restrict__ Bp,
                        float* __restrict__ C, int M, int Nc, int K) {
    const int lane = threadIdx.x & 31;
    const int wave = blockIdx.x * (blockDim.x >> 5) + (threadIdx.x >> 5);
    const int ngroups = Nc >> 6;                 // 64-wide column groups
    const int mt = wave / ngroups;
    const int ng = wave - mt * ngroups;
    if (mt * 16 >= M) return;

    const int l15  = lane & 15;
    const int half = lane >> 4;
    const int ksteps = K >> 5;

    // A 16x32 bf16 fragment: row = mt*16 + (lane&15); K = {kb..kb+15}, kb = half*16
    const unsigned short* Ap = A + (size_t)(mt * 16 + l15) * K + half * 16;
    // packed B fragments for n-tiles ng*4 .. ng*4+3
    const uint4* BpV = (const uint4*)(Bp) +
                       ((size_t)(ng * 4) * ksteps + 0) * 32 * 2;  // 2 uint4 per lane-frag

    v8f acc[4] = {};
    for (int ks = 0; ks < ksteps; ++ks) {
        FragBF a;
        a.q[0] = *(const uint4*)(Ap + ks * 32);
        a.q[1] = *(const uint4*)(Ap + ks * 32 + 8);
#pragma unroll
        for (int j = 0; j < 4; ++j) {
            FragBF b;
            const uint4* bp = BpV + ((size_t)j * ksteps + ks) * 32 * 2 + lane * 2;
            b.q[0] = bp[0];
            b.q[1] = bp[1];
            acc[j] = __builtin_amdgcn_wmma_f32_16x16x32_bf16(
                false, a.v, false, b.v, (short)0, acc[j], false, false);
        }
    }

    // C/D layout: VGPR r -> row (r + 8*half), col = lane&15
#pragma unroll
    for (int j = 0; j < 4; ++j) {
        float* Cp = C + (size_t)(mt * 16 + half * 8) * Nc + ng * 64 + j * 16 + l15;
#pragma unroll
        for (int r = 0; r < 8; ++r) Cp[(size_t)r * Nc] = acc[j][r];
    }
}

// ---------------- attention score precompute: a_src/a_dst [N,H] ----------------
__global__ void gat_att_scores(const float* __restrict__ h,
                               const float* __restrict__ att_src,
                               const float* __restrict__ att_dst,
                               float* __restrict__ asrc,
                               float* __restrict__ adst, int n) {
    int gid = blockIdx.x * blockDim.x + threadIdx.x;
    if (gid >= n * HH) return;
    int node = gid >> 2, head = gid & 3;
    const float4* hp = (const float4*)(h + (size_t)node * HC + head * CC);
    const float4* as = (const float4*)(att_src + head * CC);
    const float4* ad = (const float4*)(att_dst + head * CC);
    float ss = 0.f, sd = 0.f;
#pragma unroll
    for (int i = 0; i < CC / 4; ++i) {
        float4 hv = hp[i], av = as[i], dv = ad[i];
        ss += hv.x * av.x + hv.y * av.y + hv.z * av.z + hv.w * av.w;
        sd += hv.x * dv.x + hv.y * dv.y + hv.z * dv.z + hv.w * dv.w;
    }
    asrc[gid] = ss;
    adst[gid] = sd;
}

// ---------------- edge logits (leaky relu) + segment max ----------------
__global__ void gat_edge_logits(const float* __restrict__ asrc,
                                const float* __restrict__ adst,
                                const int* __restrict__ eidx,
                                float* __restrict__ alpha,
                                float* __restrict__ amax, int etot) {
    int e = blockIdx.x * blockDim.x + threadIdx.x;
    if (e >= etot) return;
    int s, d;
    if (e < EE) { s = eidx[e]; d = eidx[EE + e]; } else { s = d = e - EE; }
#pragma unroll
    for (int hh = 0; hh < HH; ++hh) {
        float a = asrc[s * HH + hh] + adst[d * HH + hh];
        a = a > 0.f ? a : 0.2f * a;                    // leaky relu, slope 0.2
        alpha[(size_t)e * HH + hh] = a;
        atomicMaxF(&amax[d * HH + hh], a);
    }
}

// ---------------- exp(alpha - max) + segment sum ----------------
__global__ void gat_edge_exp(float* __restrict__ alpha,
                             const float* __restrict__ amax,
                             float* __restrict__ denom,
                             const int* __restrict__ eidx, int etot) {
    int e = blockIdx.x * blockDim.x + threadIdx.x;
    if (e >= etot) return;
    int d = (e < EE) ? eidx[EE + e] : (e - EE);
#pragma unroll
    for (int hh = 0; hh < HH; ++hh) {
        float ex = __expf(alpha[(size_t)e * HH + hh] - amax[d * HH + hh]);
        alpha[(size_t)e * HH + hh] = ex;
        atomicAdd(&denom[d * HH + hh], ex);
    }
}

// ---------------- weighted message scatter: out[dst] += h[src] * w ----------------
// 64 threads per edge, each thread handles 4 contiguous channels (float4 gather).
// All 32 lanes of a wave share one edge -> index loads are wave-uniform; weight loads span 2 heads.
__global__ void gat_edge_scatter(const float* __restrict__ hfeat,
                                 const float* __restrict__ alpha,
                                 const float* __restrict__ denom,
                                 const int* __restrict__ eidx,
                                 float* __restrict__ out, int etot) {
    int gid = blockIdx.x * blockDim.x + threadIdx.x;
    int e = gid >> 6;
    if (e >= etot) return;
    int t = gid & 63;
    int c0 = t * 4;
    int hh = t >> 4;          // head = channel/64
    int s, d;
    if (e < EE) { s = eidx[e]; d = eidx[EE + e]; } else { s = d = e - EE; }
    float w = alpha[(size_t)e * HH + hh] / denom[d * HH + hh];
    float4 hv = *(const float4*)(hfeat + (size_t)s * HC + c0);
    float* op = out + (size_t)d * HC + c0;
    atomicAdd(op + 0, hv.x * w);
    atomicAdd(op + 1, hv.y * w);
    atomicAdd(op + 2, hv.z * w);
    atomicAdd(op + 3, hv.w * w);
}

// ---------------- layer1 finalize: relu(acc + b1) -> bf16 for next GEMM ----------------
__global__ void gat_finalize_concat(const float* __restrict__ acc,
                                    const float* __restrict__ bias,
                                    unsigned short* __restrict__ outbf, int n) {
    int gid = blockIdx.x * blockDim.x + threadIdx.x;
    if (gid >= n * HC) return;
    int c = gid & (HC - 1);
    float v = acc[gid] + bias[c];
    v = v > 0.f ? v : 0.f;
    outbf[gid] = f2bf(v);
}

// ---------------- layer2 finalize: relu(mean_heads(acc) + b2) -> h2 [N,64] ----------------
__global__ void gat_finalize_mean(const float* __restrict__ acc,
                                  const float* __restrict__ bias,
                                  float* __restrict__ h2, int n) {
    int gid = blockIdx.x * blockDim.x + threadIdx.x;
    if (gid >= n * CC) return;
    int node = gid >> 6, c = gid & 63;
    const float* p = acc + (size_t)node * HC + c;
    float v = 0.25f * (p[0] + p[CC] + p[2 * CC] + p[3 * CC]) + bias[c];
    h2[gid] = v > 0.f ? v : 0.f;
}

// ---------------- feed-forward head: relu(h2@ff1+b) @ ff2 + b -> [N,2] ----------------
__global__ void gat_ff_head(const float* __restrict__ h2,
                            const float* __restrict__ w1, const float* __restrict__ b1,
                            const float* __restrict__ w2, const float* __restrict__ b2,
                            float* __restrict__ out, int n) {
    int node = blockIdx.x * blockDim.x + threadIdx.x;
    if (node >= n) return;
    float in[64];
    const float4* hp = (const float4*)(h2 + (size_t)node * 64);
#pragma unroll
    for (int i = 0; i < 16; ++i) {
        float4 v = hp[i];
        in[4 * i] = v.x; in[4 * i + 1] = v.y; in[4 * i + 2] = v.z; in[4 * i + 3] = v.w;
    }
    float o0 = b2[0], o1 = b2[1];
    for (int j = 0; j < 32; ++j) {
        float s = b1[j];
#pragma unroll
        for (int k = 0; k < 64; ++k) s += in[k] * w1[k * 32 + j];  // ff1_w [64,32]
        s = s > 0.f ? s : 0.f;
        o0 += s * w2[2 * j];
        o1 += s * w2[2 * j + 1];
    }
    out[(size_t)node * 2]     = o0;
    out[(size_t)node * 2 + 1] = o1;
}

// ---------------- host side ----------------
extern "C" void kernel_launch(void* const* d_in, const int* in_sizes, int n_in,
                              void* d_out, int out_size, void* d_ws, size_t ws_size,
                              hipStream_t stream) {
    const float* x        = (const float*)d_in[0];
    const int*   eidx     = (const int*)d_in[1];
    // d_in[2] = edge_attr: ignored by reference
    const float* W1       = (const float*)d_in[3];
    const float* att_src1 = (const float*)d_in[4];
    const float* att_dst1 = (const float*)d_in[5];
    const float* b1       = (const float*)d_in[6];
    const float* W2       = (const float*)d_in[7];
    const float* att_src2 = (const float*)d_in[8];
    const float* att_dst2 = (const float*)d_in[9];
    const float* b2       = (const float*)d_in[10];
    const float* ff1w     = (const float*)d_in[11];
    const float* ff1b     = (const float*)d_in[12];
    const float* ff2w     = (const float*)d_in[13];
    const float* ff2b     = (const float*)d_in[14];
    float* out = (float*)d_out;

    // workspace carving (256B aligned)
    char* ws = (char*)d_ws;
    size_t off = 0;
    auto carve = [&](size_t bytes) -> char* {
        char* p = ws + off;
        off += (bytes + 255) & ~(size_t)255;
        return p;
    };
    unsigned short* xb    = (unsigned short*)carve((size_t)NN * FIN * 2);
    unsigned short* w1p   = (unsigned short*)carve((size_t)FIN * HC * 2);  // packed fragments
    unsigned short* w2p   = (unsigned short*)carve((size_t)HC * HC * 2);   // packed fragments
    unsigned short* h1b   = (unsigned short*)carve((size_t)NN * HC * 2);
    float* bufA  = (float*)carve((size_t)NN * HC * 4);   // GEMM output (h features)
    float* bufB  = (float*)carve((size_t)NN * HC * 4);   // scatter accumulator
    float* alpha = (float*)carve((size_t)ETOT * HH * 4);
    float* asrc  = (float*)carve((size_t)NN * HH * 4);
    float* adst  = (float*)carve((size_t)NN * HH * 4);
    float* amax  = (float*)carve((size_t)NN * HH * 4);
    float* denom = (float*)carve((size_t)NN * HH * 4);
    float* h2    = bufA;  // reuse: h2pre dead after scatter of layer 2

    const int T = 256;
    auto blocks = [](long long n, int t) { return (unsigned)((n + t - 1) / t); };
    const unsigned gemmBlocks = ((NN / 16) * (HC / 64) + 7) / 8;   // 8 waves/block
    const float NEG_BIG = -3.402823466e38f;

    // ---- bf16 conversion of x + weight fragment packing ----
    gat_f32_to_bf16<<<blocks((long long)NN * FIN, T), T, 0, stream>>>(x, xb, NN * FIN);
    gat_pack_B<<<blocks((HC / 16) * (FIN / 32) * 32, T), T, 0, stream>>>(W1, w1p, FIN, HC);
    gat_pack_B<<<blocks((HC / 16) * (HC / 32) * 32, T), T, 0, stream>>>(W2, w2p, HC, HC);

    // ================= GAT layer 1 =================
    gat_gemm_bf16_wmma<<<gemmBlocks, T, 0, stream>>>(xb, w1p, bufA, NN, HC, FIN);
    gat_att_scores<<<blocks((long long)NN * HH, T), T, 0, stream>>>(
        bufA, att_src1, att_dst1, asrc, adst, NN);
    gat_fill<<<blocks((long long)NN * HH, T), T, 0, stream>>>(amax, NEG_BIG, NN * HH);
    gat_fill<<<blocks((long long)NN * HH, T), T, 0, stream>>>(denom, 0.f, NN * HH);
    gat_fill<<<blocks((long long)NN * HC, T), T, 0, stream>>>(bufB, 0.f, NN * HC);
    gat_edge_logits<<<blocks(ETOT, T), T, 0, stream>>>(asrc, adst, eidx, alpha, amax, ETOT);
    gat_edge_exp<<<blocks(ETOT, T), T, 0, stream>>>(alpha, amax, denom, eidx, ETOT);
    gat_edge_scatter<<<blocks((long long)ETOT * 64, T), T, 0, stream>>>(
        bufA, alpha, denom, eidx, bufB, ETOT);
    gat_finalize_concat<<<blocks((long long)NN * HC, T), T, 0, stream>>>(bufB, b1, h1b, NN);

    // ================= GAT layer 2 =================
    gat_gemm_bf16_wmma<<<gemmBlocks, T, 0, stream>>>(h1b, w2p, bufA, NN, HC, HC);
    gat_att_scores<<<blocks((long long)NN * HH, T), T, 0, stream>>>(
        bufA, att_src2, att_dst2, asrc, adst, NN);
    gat_fill<<<blocks((long long)NN * HH, T), T, 0, stream>>>(amax, NEG_BIG, NN * HH);
    gat_fill<<<blocks((long long)NN * HH, T), T, 0, stream>>>(denom, 0.f, NN * HH);
    gat_fill<<<blocks((long long)NN * HC, T), T, 0, stream>>>(bufB, 0.f, NN * HC);
    gat_edge_logits<<<blocks(ETOT, T), T, 0, stream>>>(asrc, adst, eidx, alpha, amax, ETOT);
    gat_edge_exp<<<blocks(ETOT, T), T, 0, stream>>>(alpha, amax, denom, eidx, ETOT);
    gat_edge_scatter<<<blocks((long long)ETOT * 64, T), T, 0, stream>>>(
        bufA, alpha, denom, eidx, bufB, ETOT);
    gat_finalize_mean<<<blocks((long long)NN * CC, T), T, 0, stream>>>(bufB, b2, h2, NN);

    // ================= feed-forward head =================
    gat_ff_head<<<blocks(NN, T), T, 0, stream>>>(h2, ff1w, ff1b, ff2w, ff2b, out, NN);
}
